// Transformer_Block_532575945176
// MI455X (gfx1250) — matrix operand order repaired and checked by reading, not compiled, not run on hardware
//
#include <hip/hip_runtime.h>
#include <stdint.h>

// ---------------------------------------------------------------------------
// CDNA5 (gfx1250) wave32 WMMA transformer block.
// Matmuls: v_wmma_f32_16x16x32_bf16.  A-tiles: TDM tensor_load_to_lds.
// B-tiles (NT): global_load_async_to_lds_b128.  bf16 stored as raw u16.
// ---------------------------------------------------------------------------

typedef __attribute__((ext_vector_type(16))) __bf16 bf16x16;
typedef __attribute__((ext_vector_type(8)))  float  f32x8;
typedef __attribute__((ext_vector_type(4)))  unsigned int u32x4;
typedef __attribute__((ext_vector_type(8)))  int    i32x8;
typedef __attribute__((ext_vector_type(4)))  int    i32x4;

#define TBLK 128   // threads per GEMM block (4 waves)
#define TM 64
#define TN 64
#define TKS 64
#define LDT 72     // LDS row pitch in bf16 elements (144B rows, 16B aligned)

__device__ __forceinline__ unsigned short f2bf(float f) {
  union { float f; uint32_t u; } c; c.f = f;
  uint32_t u = c.u;
  uint32_t r = (u + 0x7FFFu + ((u >> 16) & 1u)) >> 16;   // RNE
  return (unsigned short)r;
}

// TDM: DMA a [rows x 64] bf16 tile (row stride = stride_elems) into LDS at
// lds_addr, padding 16B after every 128B row -> LDS pitch 72 elements (LDT).
__device__ __forceinline__ void tdm_load_tile_2d(uint32_t lds_addr,
                                                 const void* gptr,
                                                 int stride_elems) {
  uint64_t ga = (uint64_t)(uintptr_t)gptr;
  u32x4 g0 = { 1u,                                   // count=1, user descriptor
               lds_addr,                             // LDS byte address
               (unsigned)(ga & 0xFFFFFFFFu),         // global_addr[31:0]
               (unsigned)(((ga >> 32) & 0x01FFFFFFu) // global_addr[56:32]
                          | 0x80000000u) };          // type=2 ("image")
  i32x8 g1;
  g1[0] = (int)0x07110000u;        // data_size=2B, pad_en, every 128B, pad 16B
  g1[1] = 0;                       // atomic_barrier=0, tensor_dim0 lo16 (2^30)
  g1[2] = 0x4000;                  // tensor_dim0 hi16 = 2^30
  g1[3] = (TKS << 16) | 0x4000;    // tile_dim0=64 | tensor_dim1 hi16 = 2^30
  g1[4] = TM;                      // tile_dim1=64, tile_dim2=0
  g1[5] = stride_elems;            // tensor_dim0_stride[31:0]
  g1[6] = 0;
  g1[7] = 0;
  i32x4 z4 = {0, 0, 0, 0};
#if __clang_major__ >= 23
  i32x8 z8 = {0, 0, 0, 0, 0, 0, 0, 0};
  __builtin_amdgcn_tensor_load_to_lds(g0, g1, z4, z4, z8, 0);
#else
  __builtin_amdgcn_tensor_load_to_lds(g0, g1, z4, z4, 0);
#endif
}

// A fragment: 16x32 bf16 (MxK). lanes0-15: M=lane, K in {0..7,16..23};
// lanes16-31: M=lane-16, K in {8..15,24..31}. 2 bf16 per dword, low=even K.
__device__ __forceinline__ bf16x16 load_frag_a(const unsigned short* s, int r0,
                                               int k0, int lane) {
  union { bf16x16 v; uint32_t u[8]; } f;
  const unsigned short* base =
      s + (r0 + (lane & 15)) * LDT + k0 + ((lane >> 4) << 3);
#pragma unroll
  for (int i = 0; i < 8; ++i) {
    int kk = (i < 4) ? (2 * i) : (2 * i + 8);
    f.u[i] = *(const uint32_t*)(base + kk);
  }
  return f.v;
}

// B fragment: 32x16 bf16 (KxN), sourced from LDS staged as [N][K].
__device__ __forceinline__ bf16x16 load_frag_b(const unsigned short* s, int n0,
                                               int k0, int lane) {
  union { bf16x16 v; uint32_t u[8]; } f;
  const unsigned short* base =
      s + (n0 + (lane & 15)) * LDT + k0 + ((lane >> 4) << 3);
#pragma unroll
  for (int i = 0; i < 8; ++i)
    f.u[i] = *(const uint32_t*)(base + ((i < 4) ? (2 * i) : (2 * i + 8)));
  return f.v;
}

// Generic batched bf16 GEMM, C[64x64] tile per block, 4 waves of 2x2 16x16.
// bIsNT: B given as [N][K] row-major; else [K][N] (transposed while staging).
// epi: 0 = bias/resid/relu/outF/outB; 2 = causal-masked score write (-inf k>q);
//      3 = rel-shift scatter-add: scores[r][cc-1023+r] += acc (cc = j).
__global__ __launch_bounds__(TBLK) void gemm_bf16(
    const unsigned short* __restrict__ A, int lda, long long aSb, long long aSh,
    const unsigned short* __restrict__ B, int ldb, long long bSb, long long bSh,
    int bIsNT, int K, int zdiv,
    const float* __restrict__ bias,
    const float* __restrict__ resid, int resLd,
    float* __restrict__ outF, int ldF, long long fSb, long long fSh,
    unsigned short* __restrict__ outB, int ldOB, long long oSb, long long oSh,
    int relu, int epi) {
  __shared__ __align__(16) unsigned short sA[TM * LDT];
  __shared__ __align__(16) unsigned short sB[TN * LDT];

  const int tid  = threadIdx.x;
  const int lane = tid & 31;
  const int wave = tid >> 5;
  const int zb = blockIdx.z / zdiv;
  const int zh = blockIdx.z % zdiv;
  const int row0 = blockIdx.y * TM;
  const int col0 = blockIdx.x * TN;

  // BD tiles entirely below the shifted diagonal contribute nothing.
  if (epi == 3 && (row0 + col0 + (TM - 1) + (TN - 1) < 1023)) return;

  const unsigned short* Ab = A + zb * aSb + zh * aSh;
  const unsigned short* Bb = B + zb * bSb + zh * bSh;

  f32x8 acc00 = {}, acc01 = {}, acc10 = {}, acc11 = {};
  const int wr = (wave >> 1) * 32;
  const int wn = (wave & 1) * 32;

  for (int k0 = 0; k0 < K; k0 += TKS) {
    // ---- A tile [64 rows][64 K]: one TDM descriptor issued by wave 0 ----
    if (wave == 0)
      tdm_load_tile_2d((uint32_t)(uintptr_t)&sA[0],
                       Ab + (size_t)row0 * lda + k0, lda);

    // ---- B tile ----
    if (bIsNT) {  // already [N][K]: per-lane async copies straight to LDS
#pragma unroll
      for (int u = 0; u < 4; ++u) {
        int i = tid + u * TBLK;
        int r = i >> 3, c = (i & 7) << 3;
        uint32_t lo = (uint32_t)(uintptr_t)&sB[r * LDT + c];
        const unsigned short* g = Bb + (size_t)(col0 + r) * ldb + k0 + c;
        asm volatile("global_load_async_to_lds_b128 %0, %1, off"
                     :: "v"(lo), "v"(g) : "memory");
      }
    } else {      // B is [K][N]: transpose into LDS as [N][K]
      int kk0 = tid >> 3;             // 0..15
      int c0  = (tid & 7) << 3;       // 0..56
      const unsigned short* gb = Bb + (size_t)k0 * ldb + col0 + c0;
      uint4 t0 = *(const uint4*)(gb + (size_t)(kk0 +  0) * ldb);
      uint4 t1 = *(const uint4*)(gb + (size_t)(kk0 + 16) * ldb);
      uint4 t2 = *(const uint4*)(gb + (size_t)(kk0 + 32) * ldb);
      uint4 t3 = *(const uint4*)(gb + (size_t)(kk0 + 48) * ldb);
      union { uint4 q; unsigned short h[8]; } w;
#pragma unroll
      for (int u = 0; u < 4; ++u) {
        w.q = (u == 0) ? t0 : (u == 1) ? t1 : (u == 2) ? t2 : t3;
        int kk = kk0 + 16 * u;
#pragma unroll
        for (int j = 0; j < 8; ++j) sB[(c0 + j) * LDT + kk] = w.h[j];
      }
    }

    if (k0 + TKS < K)  // global_prefetch_b8 of next A tile region
      __builtin_prefetch(Ab + (size_t)(row0 + (tid >> 1)) * lda + k0 + TKS,
                         0, 1);

    __builtin_amdgcn_s_wait_tensorcnt((short)0);
    asm volatile("s_wait_asynccnt 0" ::: "memory");
    __syncthreads();

#pragma unroll
    for (int kk = 0; kk < TKS; kk += 32) {
      bf16x16 a0 = load_frag_a(sA, wr,      kk, lane);
      bf16x16 a1 = load_frag_a(sA, wr + 16, kk, lane);
      bf16x16 b0 = load_frag_b(sB, wn,      kk, lane);
      bf16x16 b1 = load_frag_b(sB, wn + 16, kk, lane);
      acc00 = __builtin_amdgcn_wmma_f32_16x16x32_bf16(false, a0, false, b0,
                                                      (short)0, acc00, false, false);
      acc01 = __builtin_amdgcn_wmma_f32_16x16x32_bf16(false, a0, false, b1,
                                                      (short)0, acc01, false, false);
      acc10 = __builtin_amdgcn_wmma_f32_16x16x32_bf16(false, a1, false, b0,
                                                      (short)0, acc10, false, false);
      acc11 = __builtin_amdgcn_wmma_f32_16x16x32_bf16(false, a1, false, b1,
                                                      (short)0, acc11, false, false);
    }
    __syncthreads();
  }

  float* outFb = outF ? outF + zb * fSb + zh * fSh : (float*)0;
  unsigned short* outBb = outB ? outB + zb * oSb + zh * oSh : (unsigned short*)0;

  f32x8 accs[2][2] = {{acc00, acc01}, {acc10, acc11}};
#pragma unroll
  for (int ti = 0; ti < 2; ++ti)
#pragma unroll
    for (int tj = 0; tj < 2; ++tj) {
      f32x8 c = accs[ti][tj];
      int rb = row0 + wr + ti * 16 + ((lane >> 4) << 3);
      int cc = col0 + wn + tj * 16 + (lane & 15);
#pragma unroll
      for (int i = 0; i < 8; ++i) {
        int r = rb + i;
        float v = c[i];
        if (epi == 2) {
          outFb[(size_t)r * ldF + cc] = (cc <= r) ? v : -__builtin_inff();
        } else if (epi == 3) {
          int k = cc - 1023 + r;  // rel-shift: unique writer per (r,k)
          if (k >= 0) outFb[(size_t)r * ldF + k] += v;
        } else {
          if (bias)  v += bias[cc];
          if (resid) v += resid[(size_t)r * resLd + cc];
          if (relu)  v = fmaxf(v, 0.0f);
          if (outFb) outFb[(size_t)r * ldF + cc] = v;
          if (outBb) outBb[(size_t)r * ldOB + cc] = f2bf(v);
        }
      }
    }
}

// Row softmax over 1024 keys, scale 1/8 folded in; writes bf16 probs in place
// over the fp32 row (all reads complete before the reduction barriers).
__global__ __launch_bounds__(256) void softmax_bf16_inplace(
    float* __restrict__ scores) {
  float* row = scores + (size_t)blockIdx.x * 1024;
  int t = threadIdx.x;
  float4 x = ((const float4*)row)[t];
  __shared__ float red[256];
  red[t] = fmaxf(fmaxf(x.x, x.y), fmaxf(x.z, x.w));
  __syncthreads();
  for (int s = 128; s > 0; s >>= 1) {
    if (t < s) red[t] = fmaxf(red[t], red[t + s]);
    __syncthreads();
  }
  float m = red[0];
  __syncthreads();
  const float sc = 0.125f;  // 1/sqrt(64)
  float e0 = expf((x.x - m) * sc), e1 = expf((x.y - m) * sc);
  float e2 = expf((x.z - m) * sc), e3 = expf((x.w - m) * sc);
  red[t] = e0 + e1 + e2 + e3;
  __syncthreads();
  for (int s = 128; s > 0; s >>= 1) {
    if (t < s) red[t] += red[t + s];
    __syncthreads();
  }
  float inv = 1.0f / red[0];
  __syncthreads();
  unsigned short* p = (unsigned short*)row;
  p[4 * t + 0] = f2bf(e0 * inv);
  p[4 * t + 1] = f2bf(e1 * inv);
  p[4 * t + 2] = f2bf(e2 * inv);
  p[4 * t + 3] = f2bf(e3 * inv);
}

__global__ __launch_bounds__(256) void layernorm_1024(
    const float* __restrict__ in, const float* __restrict__ g,
    const float* __restrict__ b, float* __restrict__ outF,
    unsigned short* __restrict__ outB) {
  const float* row = in + (size_t)blockIdx.x * 1024;
  int t = threadIdx.x;
  float4 x = ((const float4*)row)[t];
  __shared__ float red[256];
  red[t] = x.x + x.y + x.z + x.w;
  __syncthreads();
  for (int s = 128; s > 0; s >>= 1) {
    if (t < s) red[t] += red[t + s];
    __syncthreads();
  }
  float mean = red[0] * (1.0f / 1024.0f);
  __syncthreads();
  float d0 = x.x - mean, d1 = x.y - mean, d2 = x.z - mean, d3 = x.w - mean;
  red[t] = d0 * d0 + d1 * d1 + d2 * d2 + d3 * d3;
  __syncthreads();
  for (int s = 128; s > 0; s >>= 1) {
    if (t < s) red[t] += red[t + s];
    __syncthreads();
  }
  float rstd = rsqrtf(red[0] * (1.0f / 1024.0f) + 1e-5f);
  float4 gv = ((const float4*)g)[t];
  float4 bv = ((const float4*)b)[t];
  float y0 = d0 * rstd * gv.x + bv.x, y1 = d1 * rstd * gv.y + bv.y;
  float y2 = d2 * rstd * gv.z + bv.z, y3 = d3 * rstd * gv.w + bv.w;
  size_t o = (size_t)blockIdx.x * 1024 + 4 * t;
  if (outF) { outF[o] = y0; outF[o + 1] = y1; outF[o + 2] = y2; outF[o + 3] = y3; }
  if (outB) { outB[o] = f2bf(y0); outB[o + 1] = f2bf(y1);
              outB[o + 2] = f2bf(y2); outB[o + 3] = f2bf(y3); }
}

__global__ void f32_to_bf16_kernel(const float* __restrict__ in,
                                   unsigned short* __restrict__ out,
                                   long long n) {
  long long i = (long long)blockIdx.x * blockDim.x + threadIdx.x;
  long long st = (long long)gridDim.x * blockDim.x;
  for (; i < n; i += st) out[i] = f2bf(in[i]);
}

// ---------------------------------------------------------------------------
extern "C" void kernel_launch(void* const* d_in, const int* in_sizes, int n_in,
                              void* d_out, int out_size, void* d_ws,
                              size_t ws_size, hipStream_t stream) {
  (void)in_sizes; (void)n_in; (void)out_size; (void)ws_size;
  const float* x    = (const float*)d_in[0];
  const float* pos  = (const float*)d_in[2];
  const float* rr   = (const float*)d_in[3];
  const float* rw   = (const float*)d_in[4];
  const float* Wkv  = (const float*)d_in[5];
  const float* Wq   = (const float*)d_in[6];
  const float* Wr   = (const float*)d_in[7];
  const float* Wo   = (const float*)d_in[8];
  const float* ln1g = (const float*)d_in[9];
  const float* ln1b = (const float*)d_in[10];
  const float* W1   = (const float*)d_in[11];
  const float* b1   = (const float*)d_in[12];
  const float* W2   = (const float*)d_in[13];
  const float* b2   = (const float*)d_in[14];
  const float* ln2g = (const float*)d_in[15];
  const float* ln2b = (const float*)d_in[16];

  typedef unsigned short bf;
  char* base = (char*)d_ws;
  size_t off = 0;
  auto alloc = [&](size_t bytes) -> void* {
    void* p = base + off;
    off += (bytes + 255) & ~(size_t)255;
    return p;
  };
  bf*    xb    = (bf*)alloc(2048ll * 1024 * 2);
  bf*    posb  = (bf*)alloc(1024ll * 1024 * 2);
  bf*    Wqb   = (bf*)alloc(1024ll * 1024 * 2);
  bf*    Wkvb  = (bf*)alloc(1024ll * 2048 * 2);
  bf*    Wrb   = (bf*)alloc(1024ll * 1024 * 2);
  bf*    Wob   = (bf*)alloc(1024ll * 1024 * 2);
  bf*    W1b   = (bf*)alloc(1024ll * 4096 * 2);
  bf*    W2b   = (bf*)alloc(4096ll * 1024 * 2);
  bf*    qrw   = (bf*)alloc(2048ll * 1024 * 2);
  bf*    qrr   = (bf*)alloc(2048ll * 1024 * 2);
  bf*    kvb   = (bf*)alloc(2048ll * 2048 * 2);
  bf*    rb    = (bf*)alloc(1024ll * 1024 * 2);
  float* scores= (float*)alloc(2ll * 16 * 1024 * 1024 * 4);  // 128 MB
  bf*    att   = (bf*)alloc(2048ll * 1024 * 2);
  float* pre1  = (float*)alloc(2048ll * 1024 * 4);
  float* out1f = (float*)alloc(2048ll * 1024 * 4);
  bf*    out1b = (bf*)alloc(2048ll * 1024 * 2);
  bf*    hb    = (bf*)alloc(2048ll * 4096 * 2);
  float* pre2  = (float*)alloc(2048ll * 1024 * 4);

  // 1) fp32 -> bf16 converts
  f32_to_bf16_kernel<<<1024, 256, 0, stream>>>(x,   xb,   2048ll * 1024);
  f32_to_bf16_kernel<<<1024, 256, 0, stream>>>(pos, posb, 1024ll * 1024);
  f32_to_bf16_kernel<<<1024, 256, 0, stream>>>(Wq,  Wqb,  1024ll * 1024);
  f32_to_bf16_kernel<<<1024, 256, 0, stream>>>(Wkv, Wkvb, 1024ll * 2048);
  f32_to_bf16_kernel<<<1024, 256, 0, stream>>>(Wr,  Wrb,  1024ll * 1024);
  f32_to_bf16_kernel<<<1024, 256, 0, stream>>>(Wo,  Wob,  1024ll * 1024);
  f32_to_bf16_kernel<<<1024, 256, 0, stream>>>(W1,  W1b,  1024ll * 4096);
  f32_to_bf16_kernel<<<1024, 256, 0, stream>>>(W2,  W2b,  4096ll * 1024);

  // 2) projections: qrw = x@Wq + rw, qrr = x@Wq + rr, kv = x@Wkv, r = pos@Wr
  gemm_bf16<<<dim3(16, 32, 1), TBLK, 0, stream>>>(
      xb, 1024, 0, 0, Wqb, 1024, 0, 0, 0, 1024, 1,
      rw, nullptr, 0, nullptr, 0, 0, 0, qrw, 1024, 0, 0, 0, 0);
  gemm_bf16<<<dim3(16, 32, 1), TBLK, 0, stream>>>(
      xb, 1024, 0, 0, Wqb, 1024, 0, 0, 0, 1024, 1,
      rr, nullptr, 0, nullptr, 0, 0, 0, qrr, 1024, 0, 0, 0, 0);
  gemm_bf16<<<dim3(32, 32, 1), TBLK, 0, stream>>>(
      xb, 1024, 0, 0, Wkvb, 2048, 0, 0, 0, 1024, 1,
      nullptr, nullptr, 0, nullptr, 0, 0, 0, kvb, 2048, 0, 0, 0, 0);
  gemm_bf16<<<dim3(16, 16, 1), TBLK, 0, stream>>>(
      posb, 1024, 0, 0, Wrb, 1024, 0, 0, 0, 1024, 1,
      nullptr, nullptr, 0, nullptr, 0, 0, 0, rb, 1024, 0, 0, 0, 0);

  // 3) AC scores: per (b,h) qrw_bh @ k_bh^T, causal -inf for k>q  (epi=2)
  gemm_bf16<<<dim3(16, 16, 32), TBLK, 0, stream>>>(
      qrw, 1024, 1024ll * 1024, 64,
      kvb, 2048, 1024ll * 2048, 64, 1, 64, 16,
      nullptr, nullptr, 0,
      scores, 1024, 16ll * 1024 * 1024, 1024ll * 1024,
      nullptr, 0, 0, 0, 0, 2);
  // 4) BD: per (b,h) qrr_bh @ r_h^T with rel-shift scatter-add  (epi=3)
  gemm_bf16<<<dim3(16, 16, 32), TBLK, 0, stream>>>(
      qrr, 1024, 1024ll * 1024, 64,
      rb, 1024, 0, 64, 1, 64, 16,
      nullptr, nullptr, 0,
      scores, 1024, 16ll * 1024 * 1024, 1024ll * 1024,
      nullptr, 0, 0, 0, 0, 3);

  // 5) softmax rows -> bf16 probs in place (pitch 2048 bf16 per row)
  softmax_bf16_inplace<<<2 * 16 * 1024, 256, 0, stream>>>(scores);

  // 6) att = prob @ v  (batched over b,h; N=64)
  gemm_bf16<<<dim3(1, 16, 32), TBLK, 0, stream>>>(
      (bf*)scores, 2048, 16ll * 1024 * 2048, 1024ll * 2048,
      kvb + 1024, 2048, 1024ll * 2048, 64, 0, 1024, 16,
      nullptr, nullptr, 0, nullptr, 0, 0, 0,
      att, 1024, 1024ll * 1024, 64, 0, 0);

  // 7) pre1 = x + att@Wo ; LN1
  gemm_bf16<<<dim3(16, 32, 1), TBLK, 0, stream>>>(
      att, 1024, 0, 0, Wob, 1024, 0, 0, 0, 1024, 1,
      nullptr, x, 1024, pre1, 1024, 0, 0, nullptr, 0, 0, 0, 0, 0);
  layernorm_1024<<<2048, 256, 0, stream>>>(pre1, ln1g, ln1b, out1f, out1b);

  // 8) FFN: h = relu(out1@W1 + b1); pre2 = out1 + h@W2 + b2; LN2 -> d_out
  gemm_bf16<<<dim3(64, 32, 1), TBLK, 0, stream>>>(
      out1b, 1024, 0, 0, W1b, 4096, 0, 0, 0, 1024, 1,
      b1, nullptr, 0, nullptr, 0, 0, 0, hb, 4096, 0, 0, 1, 0);
  gemm_bf16<<<dim3(16, 32, 1), TBLK, 0, stream>>>(
      hb, 4096, 0, 0, W2b, 1024, 0, 0, 0, 4096, 1,
      b2, out1f, 1024, pre2, 1024, 0, 0, nullptr, 0, 0, 0, 0, 0);
  layernorm_1024<<<2048, 256, 0, stream>>>(pre2, ln2g, ln2b, (float*)d_out,
                                           nullptr);
}
